// GMMAttention_47845935677727
// MI455X (gfx1250) — compile-verified, matrix operand order checked by reading, not run on hardware
//
#include <hip/hip_runtime.h>
#include <math.h>
#include <stdint.h>

#define Bb   64
#define Tt   2048
#define Ee   512
#define RNN  1024
#define ATT  128
#define Kk   8

typedef __attribute__((ext_vector_type(2)))  float    v2f;
typedef __attribute__((ext_vector_type(8)))  float    v8f;
typedef __attribute__((ext_vector_type(16))) _Float16 v16h;

// ---------------------------------------------------------------------------
// Kernel 1: h = tanh(query @ W1^T + b1), 64x1024 @ 1024x128 -> 64x128, WMMA.
// Grid (4 m-tiles, 8 n-tiles), one wave32 per block (EXEC all ones for WMMA).
// f32 A 16x4 layout: lane<16 -> m=lane, K={0,1}; lane>=16 -> m=lane-16, K={2,3}
// f32 B 4x16 layout (symmetric): lane<16 -> n=lane, K={0,1}; lane>=16 K={2,3}
// D 16x16 f32: VGPR v, lane L -> M = v + 8*(L>=16), N = L&15.
// ---------------------------------------------------------------------------
__global__ __launch_bounds__(32)
void h_gemm_wmma(const float* __restrict__ q, const float* __restrict__ W1,
                 const float* __restrict__ b1, float* __restrict__ h)
{
    const int m0   = blockIdx.x * 16;
    const int n0   = blockIdx.y * 16;
    const int lane = threadIdx.x & 31;
    const int half = lane >> 4;     // 0: K-lo half, 1: K-hi half
    const int idx  = lane & 15;     // m for A-frag, n for B-frag

    const float* __restrict__ qrow = q  + (size_t)(m0 + idx) * RNN;
    const float* __restrict__ wrow = W1 + (size_t)(n0 + idx) * RNN; // B[k,n]=W1[n,k]

    v8f c = {};
#if __has_builtin(__builtin_amdgcn_wmma_f32_16x16x4_f32)
    const int kb = half * 2;
    for (int k0 = 0; k0 < RNN; k0 += 4) {
        v2f a, bb;
        a.x  = qrow[k0 + kb]; a.y  = qrow[k0 + kb + 1];
        bb.x = wrow[k0 + kb]; bb.y = wrow[k0 + kb + 1];
        c = __builtin_amdgcn_wmma_f32_16x16x4_f32(false, a, false, bb,
                                                  (short)0, c, false, false);
    }
#else
    // Fallback: f16 inputs, f32 accumulate (codegen-confirmed builtin).
    // 16-bit A 16x32 layout: lane<16 K = [k0..k0+7] U [k0+16..k0+23];
    // lanes>=16 shifted by 8. B symmetric.
    for (int k0 = 0; k0 < RNN; k0 += 32) {
        v16h a, bb;
        const int base = k0 + half * 8;
        #pragma unroll
        for (int i = 0; i < 8; ++i) {
            a[i]      = (_Float16)qrow[base + i];
            a[8 + i]  = (_Float16)qrow[base + 16 + i];
            bb[i]     = (_Float16)wrow[base + i];
            bb[8 + i] = (_Float16)wrow[base + 16 + i];
        }
        c = __builtin_amdgcn_wmma_f32_16x16x32_f16(false, a, false, bb,
                                                   (short)0, c, false, false);
    }
#endif
    const int n     = n0 + idx;
    const float bia = b1[n];
    #pragma unroll
    for (int v = 0; v < 8; ++v) {
        const int m = m0 + v + 8 * half;
        h[m * ATT + n] = tanhf(c[v] + bia);
    }
}

// ---------------------------------------------------------------------------
// Kernel 2: p = h @ W2^T + b2 (64x24), then per-(b,k) GMM params:
//   coef = softmax(omega_hat)/Z,  mu = mu_prev + exp(delta_hat),
//   inv  = 1/(2 sigma^2),  sigma = sqrt(exp(sigma_hat)) + 1e-5
// One block per batch row, 128 threads.
// ---------------------------------------------------------------------------
__global__ __launch_bounds__(ATT)
void params_kernel(const float* __restrict__ h, const float* __restrict__ W2,
                   const float* __restrict__ b2, const float* __restrict__ mu_prev,
                   float* __restrict__ pcoef, float* __restrict__ pmu,
                   float* __restrict__ pinv)
{
    const int b   = blockIdx.x;
    const int tid = threadIdx.x;
    __shared__ float sh[ATT];
    __shared__ float sp[3 * Kk];

    sh[tid] = h[b * ATT + tid];
    __syncthreads();

    if (tid < 3 * Kk) {
        float acc = b2[tid];
        const float* __restrict__ w = W2 + tid * ATT;
        #pragma unroll 8
        for (int i = 0; i < ATT; ++i) acc += w[i] * sh[i];
        sp[tid] = acc;
    }
    __syncthreads();

    if (tid < Kk) {
        float mx = sp[0];
        #pragma unroll
        for (int i = 1; i < Kk; ++i) mx = fmaxf(mx, sp[i]);
        float sum = 0.f;
        #pragma unroll
        for (int i = 0; i < Kk; ++i) sum += expf(sp[i] - mx);
        const float omega = expf(sp[tid] - mx) / sum;
        const float delta = expf(sp[Kk + tid]);
        const float sigma = sqrtf(expf(sp[2 * Kk + tid])) + 1e-5f;
        const float Z     = sqrtf(6.283185307179586f * sigma * sigma);
        pcoef[b * Kk + tid] = omega / Z;
        pmu  [b * Kk + tid] = mu_prev[b * Kk + tid] + delta;   // mu_prev (B,K,1)
        pinv [b * Kk + tid] = 1.f / (2.f * sigma * sigma);
    }
}

// ---------------------------------------------------------------------------
// Kernel 3: align[b,t] = sum_k coef*exp(-(t-mu)^2*inv); mask -> 1e-8.
// Grid (T/256, B), 256 threads.
// ---------------------------------------------------------------------------
__global__ __launch_bounds__(256)
void align_kernel(const float* __restrict__ pcoef, const float* __restrict__ pmu,
                  const float* __restrict__ pinv, const uint8_t* __restrict__ mask,
                  float* __restrict__ align_out)
{
    const int b = blockIdx.y;
    const int t = blockIdx.x * 256 + threadIdx.x;
    __shared__ float sc[Kk], sm[Kk], si[Kk];
    if (threadIdx.x < Kk) {
        sc[threadIdx.x] = pcoef[b * Kk + threadIdx.x];
        sm[threadIdx.x] = pmu  [b * Kk + threadIdx.x];
        si[threadIdx.x] = pinv [b * Kk + threadIdx.x];
    }
    __syncthreads();

    const float tf = (float)t;
    float acc = 0.f;
    #pragma unroll
    for (int k = 0; k < Kk; ++k) {
        const float d = tf - sm[k];
        acc += sc[k] * expf(-d * d * si[k]);
    }
    if (mask[b * Tt + t]) acc = 1e-8f;
    align_out[b * Tt + t] = acc;
}

// ---------------------------------------------------------------------------
// Kernel 4: context partials. Grid (B, NC t-chunks), 128 threads; each thread
// owns 4 consecutive e (float4 => global_load_b128), block covers full E row.
// Streams the 256 MB `memory` tensor once — this is the bandwidth-bound core.
// ---------------------------------------------------------------------------
__global__ __launch_bounds__(128)
void ctx_partial(const float* __restrict__ mem, const float* __restrict__ align,
                 float* __restrict__ part, int tc_len)
{
    const int b   = blockIdx.x;
    const int c   = blockIdx.y;
    const int tid = threadIdx.x;           // 0..127, e = tid*4
    const int t0  = c * tc_len;

    __shared__ float sa[Tt];               // up to full row (8 KB of 320 KB LDS)
    for (int i = tid; i < tc_len; i += 128) sa[i] = align[b * Tt + t0 + i];
    __syncthreads();

    const float4* __restrict__ m4 =
        reinterpret_cast<const float4*>(mem + ((size_t)b * Tt + t0) * Ee);

    float ax = 0.f, ay = 0.f, az = 0.f, aw = 0.f;
    for (int t = 0; t < tc_len; ++t) {
        if (t + 16 < tc_len)               // gfx1250 global_prefetch_b8
            __builtin_prefetch((const void*)&m4[(size_t)(t + 16) * (Ee / 4) + tid], 0, 1);
        const float  a = sa[t];
        const float4 v = m4[(size_t)t * (Ee / 4) + tid];
        ax += a * v.x; ay += a * v.y; az += a * v.z; aw += a * v.w;
    }
    float4* __restrict__ p4 =
        reinterpret_cast<float4*>(part + ((size_t)c * Bb + b) * Ee);
    float4 r; r.x = ax; r.y = ay; r.z = az; r.w = aw;
    p4[tid] = r;
}

// Kernel 5: deterministic reduction of NC t-chunk partials.
__global__ __launch_bounds__(256)
void ctx_reduce(const float* __restrict__ part, float* __restrict__ out, int NC)
{
    const int i = blockIdx.x * 256 + threadIdx.x;     // B*E = 32768
    float acc = 0.f;
    for (int c = 0; c < NC; ++c) acc += part[(size_t)c * Bb * Ee + i];
    out[i] = acc;
}

// ---------------------------------------------------------------------------
extern "C" void kernel_launch(void* const* d_in, const int* in_sizes, int n_in,
                              void* d_out, int out_size, void* d_ws, size_t ws_size,
                              hipStream_t stream)
{
    const float*   q       = (const float*)d_in[0];
    const float*   mem     = (const float*)d_in[1];
    // d_in[2] processed_memory, d_in[3] attention_weights_cat: UNUSED by reference
    const uint8_t* mask    = (const uint8_t*)d_in[4];   // jax bool_ = 1 byte
    const float*   W1      = (const float*)d_in[5];
    const float*   b1      = (const float*)d_in[6];
    const float*   W2      = (const float*)d_in[7];
    const float*   b2      = (const float*)d_in[8];
    const float*   mu_prev = (const float*)d_in[9];

    float* out_ctx   = (float*)d_out;            // (B,E) first
    float* out_align = (float*)d_out + Bb * Ee;  // then (B,T)

    char*  ws    = (char*)d_ws;
    float* h     = (float*)ws;                        // B*ATT  = 32 KB
    float* pcoef = (float*)(ws + 32768);              // B*K
    float* pmu   = pcoef + Bb * Kk;
    float* pinv  = pmu   + Bb * Kk;
    const size_t part_off = 40960;
    float* part  = (float*)(ws + part_off);

    // Pick largest power-of-two T-chunk count whose partials fit in ws.
    int NC = 1;
    const int cands[5] = {16, 8, 4, 2, 1};
    for (int i = 0; i < 5; ++i) {
        if (part_off + (size_t)cands[i] * Bb * Ee * sizeof(float) <= ws_size) {
            NC = cands[i];
            break;
        }
    }

    h_gemm_wmma  <<<dim3(4, 8),        32,  0, stream>>>(q, W1, b1, h);
    params_kernel<<<Bb,                ATT, 0, stream>>>(h, W2, b2, mu_prev,
                                                         pcoef, pmu, pinv);
    align_kernel <<<dim3(Tt / 256, Bb), 256, 0, stream>>>(pcoef, pmu, pinv,
                                                          mask, out_align);
    if (NC == 1) {
        ctx_partial<<<dim3(Bb, 1),  128, 0, stream>>>(mem, out_align, out_ctx, Tt);
    } else {
        ctx_partial<<<dim3(Bb, NC), 128, 0, stream>>>(mem, out_align, part, Tt / NC);
        ctx_reduce <<<(Bb * Ee) / 256, 256, 0, stream>>>(part, out_ctx, NC);
    }
}
